// FCLSTM_73873437491686
// MI455X (gfx1250) — matrix operand order repaired and checked by reading, not compiled, not run on hardware
//
#include <hip/hip_runtime.h>
#include <hip/hip_bf16.h>

typedef __bf16 bf16;
typedef __attribute__((ext_vector_type(16))) __bf16 v16bf;
typedef __attribute__((ext_vector_type(8)))  __bf16 bf16x8;
typedef __attribute__((ext_vector_type(8)))  float  v8f;

union AFrag { v16bf v; bf16x8 h[2]; };

#define SEQ 128
#define BATCH 128
#define HID 512

struct FArgs {
    bf16*        wcat[3][2][2];   // [tower][layer][dir] -> (2048 x Kcat) bf16, row-major
    int          kcat[3][2];
    bf16*        xbf[3];          // (B, S, Xpad) bf16
    int          xpad[3];
    bf16*        h0[3][2][2];     // [tower][dir][pingpong] (B, 512) bf16
    bf16*        h1[3][2][2];
    float*       c0[3][2];        // (B, 512) f32
    float*       c1[3][2];
    float*       st[3];           // running sum of h1[dir=1]
    const float* bih[3][2][2];
    const float* bhh[3][2][2];
};

// ---- prep: pack [Wih_x(pad) | Wih_cross | Whh] row-major into bf16 ----
__global__ void pack_w_kernel(bf16* dst, const float* wih, const float* whh,
                              int X, int Xpad, int inW, int K, int total) {
    int i = blockIdx.x * blockDim.x + threadIdx.x;
    if (i >= total) return;
    int n = i / K, k = i - n * K;
    float v;
    if (k < Xpad)             v = (k < X) ? wih[(size_t)n * inW + k] : 0.0f;
    else if (k < Xpad + 1024) v = wih[(size_t)n * inW + X + (k - Xpad)];
    else                      v = whh[(size_t)n * HID + (k - Xpad - 1024)];
    dst[i] = (bf16)v;
}

__global__ void cvt_x_kernel(bf16* dst, const float* src, int X, int Xpad, int total) {
    int i = blockIdx.x * blockDim.x + threadIdx.x;
    if (i >= total) return;
    int bs = i / Xpad, k = i - bs * Xpad;
    dst[i] = (bf16)((k < X) ? src[(size_t)bs * X + k] : 0.0f);
}

// Accumulate one K-segment of the 4 gate tiles (i,f,g,o) owned by this wave.
//   ap     : A-segment base (rows = batch, row stride rs, lenK bf16 cols)
//   W      : packed weight base; gate row n covers K at (n*Kstride + wkbase + k)
__device__ __forceinline__ void gemm_seg(const bf16* __restrict__ ap, size_t rs, int lenK,
                                         const bf16* __restrict__ W, int Kstride, int wkbase,
                                         int rbase, int lr, bool hi, int n0,
                                         v8f& a0, v8f& a1, v8f& a2, v8f& a3) {
    const bf16* arow = ap + (size_t)(rbase + lr) * rs + (hi ? 16 : 0);
    const bf16* wb   = W + (size_t)wkbase + (hi ? 16 : 0);
    const size_t w0 = (size_t)(n0)        * Kstride;
    const size_t w1 = (size_t)(n0 +  512) * Kstride;
    const size_t w2 = (size_t)(n0 + 1024) * Kstride;
    const size_t w3 = (size_t)(n0 + 1536) * Kstride;
#pragma unroll 4
    for (int k = 0; k < lenK; k += 32) {
        AFrag a;
        a.h[0] = *(const bf16x8*)(arow + k);
        a.h[1] = *(const bf16x8*)(arow + k + 8);
        v16bf bb0 = *(const v16bf*)(wb + w0 + k);
        v16bf bb1 = *(const v16bf*)(wb + w1 + k);
        v16bf bb2 = *(const v16bf*)(wb + w2 + k);
        v16bf bb3 = *(const v16bf*)(wb + w3 + k);
        a0 = __builtin_amdgcn_wmma_f32_16x16x32_bf16(false, a.v, false, bb0, (short)0, a0, false, false);
        a1 = __builtin_amdgcn_wmma_f32_16x16x32_bf16(false, a.v, false, bb1, (short)0, a1, false, false);
        a2 = __builtin_amdgcn_wmma_f32_16x16x32_bf16(false, a.v, false, bb2, (short)0, a2, false, false);
        a3 = __builtin_amdgcn_wmma_f32_16x16x32_bf16(false, a.v, false, bb3, (short)0, a3, false, false);
    }
}

// ---- one LSTM layer-stage for all 3 towers x 2 dirs, one timestep ----
// grid: (32, 3, 2)  block: 256
//   blockIdx.x: rblk = bx&7 (batch rows /16), jblk = bx>>3 (hidden cols /128)
//   blockIdx.y: tower, blockIdx.z: dir
__global__ __launch_bounds__(256) void lstm_stage(FArgs A, int s, int layer, int rp, int wp) {
    const int m    = blockIdx.y;
    const int d    = blockIdx.z;
    const int rblk = blockIdx.x & 7;
    const int jblk = blockIdx.x >> 3;
    const int wv   = threadIdx.x >> 5;
    const int lane = threadIdx.x & 31;
    const bool hi  = lane >= 16;
    const int  lr  = lane & 15;
    const int rbase = rblk * 16;
    const int jt    = jblk * 128 + wv * 16;     // hidden-col tile base
    const int K     = A.kcat[m][layer];
    const bf16* __restrict__ W = A.wcat[m][layer][d];
    const int n0 = jt + lr;

    v8f a0 = {}, a1 = {}, a2 = {}, a3 = {};

    if (layer == 0) {
        const int xp = A.xpad[m];
        const int c1 = (m == 0) ? 1 : 0;
        const int c2 = (m == 2) ? 1 : 2;
        // x (this step), then cross-feeds + recurrent h from previous step
        gemm_seg(A.xbf[m] + (size_t)s * xp, (size_t)SEQ * xp, xp,
                 W, K, 0,          rbase, lr, hi, n0, a0, a1, a2, a3);
        gemm_seg(A.h1[c1][1][rp], HID, HID,
                 W, K, xp,         rbase, lr, hi, n0, a0, a1, a2, a3);
        gemm_seg(A.h1[c2][1][rp], HID, HID,
                 W, K, xp + 512,   rbase, lr, hi, n0, a0, a1, a2, a3);
        gemm_seg(A.h0[m][d][rp],  HID, HID,
                 W, K, xp + 1024,  rbase, lr, hi, n0, a0, a1, a2, a3);
    } else {
        // layer-0 outputs of this step, then recurrent h from previous step
        gemm_seg(A.h0[m][0][wp], HID, HID, W, K, 0,    rbase, lr, hi, n0, a0, a1, a2, a3);
        gemm_seg(A.h0[m][1][wp], HID, HID, W, K, 512,  rbase, lr, hi, n0, a0, a1, a2, a3);
        gemm_seg(A.h1[m][d][rp], HID, HID, W, K, 1024, rbase, lr, hi, n0, a0, a1, a2, a3);
    }

    // fused LSTM cell elementwise; each lane owns col j, rows e+{0,8}
    const int j = jt + lr;
    const float* bi = A.bih[m][layer][d];
    const float* bh = A.bhh[m][layer][d];
    const float bsi = bi[j]        + bh[j];
    const float bsf = bi[512 + j]  + bh[512 + j];
    const float bsg = bi[1024 + j] + bh[1024 + j];
    const float bso = bi[1536 + j] + bh[1536 + j];

    float* C;
    bf16*  Hh;
    float* ST = nullptr;
    if (layer == 0) { C = A.c0[m][d]; Hh = A.h0[m][d][wp]; }
    else            { C = A.c1[m][d]; Hh = A.h1[m][d][wp]; if (d == 1) ST = A.st[m]; }

    const int rowoff = hi ? 8 : 0;
#pragma unroll
    for (int e = 0; e < 8; ++e) {
        const int row = rbase + rowoff + e;
        const size_t idx = (size_t)row * HID + j;
        const float gi = a0[e] + bsi;
        const float gf = a1[e] + bsf;
        const float gg = a2[e] + bsg;
        const float go = a3[e] + bso;
        const float iv = 1.0f / (1.0f + __expf(-gi));
        const float fv = 1.0f / (1.0f + __expf(-gf));
        const float gv = tanhf(gg);
        const float ov = 1.0f / (1.0f + __expf(-go));
        const float cn = fv * C[idx] + iv * gv;
        const float hn = ov * tanhf(cn);
        C[idx]  = cn;
        Hh[idx] = (bf16)hn;
        if (ST) ST[idx] += hn;
    }
}

__global__ void decode_kernel(const float* st0, const float* st1, const float* st2,
                              const float* decW, const float* decb, float* out) {
    int b = threadIdx.x;
    if (b >= BATCH) return;
    float s = 0.0f;
    for (int j = 0; j < HID; ++j) s += st0[(size_t)b * HID + j] * decW[j];
    for (int j = 0; j < HID; ++j) s += st1[(size_t)b * HID + j] * decW[512 + j];
    for (int j = 0; j < HID; ++j) s += st2[(size_t)b * HID + j] * decW[1024 + j];
    out[b] = s * (1.0f / (float)SEQ) + decb[0];
}

extern "C" void kernel_launch(void* const* d_in, const int* in_sizes, int n_in,
                              void* d_out, int out_size, void* d_ws, size_t ws_size,
                              hipStream_t stream) {
    (void)in_sizes; (void)out_size; (void)ws_size;
    const int X[3]  = {300, 512, 128};
    const int XP[3] = {320, 512, 128};
    int K0[3];
    for (int m = 0; m < 3; ++m) K0[m] = XP[m] + 1536;

    // flattened params: d_in[3 + m*16 + l*8 + d*4 + {Wih,Whh,bih,bhh}]
    const float* P[3][2][2][4];
    for (int m = 0; m < 3; ++m)
        for (int l = 0; l < 2; ++l)
            for (int d = 0; d < 2; ++d)
                for (int i = 0; i < 4; ++i)
                    P[m][l][d][i] = (const float*)d_in[3 + m * 16 + l * 8 + d * 4 + i];
    const float* decW = (const float*)d_in[n_in - 2];
    const float* decb = (const float*)d_in[n_in - 1];

    char* ws = (char*)d_ws;
    size_t off = 0;
    auto take = [&](size_t bytes) -> size_t {
        size_t o = off;
        off = (off + bytes + 255) & ~(size_t)255;
        return o;
    };

    FArgs A;
    const size_t hB = (size_t)BATCH * HID * sizeof(bf16);
    const size_t cB = (size_t)BATCH * HID * sizeof(float);
    // --- states first (one contiguous memset region) ---
    for (int m = 0; m < 3; ++m)
        for (int d = 0; d < 2; ++d)
            for (int p = 0; p < 2; ++p) A.h0[m][d][p] = (bf16*)(ws + take(hB));
    for (int m = 0; m < 3; ++m)
        for (int d = 0; d < 2; ++d)
            for (int p = 0; p < 2; ++p) A.h1[m][d][p] = (bf16*)(ws + take(hB));
    for (int m = 0; m < 3; ++m)
        for (int d = 0; d < 2; ++d) A.c0[m][d] = (float*)(ws + take(cB));
    for (int m = 0; m < 3; ++m)
        for (int d = 0; d < 2; ++d) A.c1[m][d] = (float*)(ws + take(cB));
    for (int m = 0; m < 3; ++m) A.st[m] = (float*)(ws + take(cB));
    const size_t stateEnd = off;
    // --- packed weights + bf16 inputs ---
    for (int m = 0; m < 3; ++m) {
        A.kcat[m][0] = K0[m];
        A.kcat[m][1] = 1536;
        for (int l = 0; l < 2; ++l)
            for (int d = 0; d < 2; ++d) {
                size_t kk = (l == 0) ? (size_t)K0[m] : 1536;
                A.wcat[m][l][d] = (bf16*)(ws + take((size_t)2048 * kk * sizeof(bf16)));
            }
    }
    for (int m = 0; m < 3; ++m) {
        A.xpad[m] = XP[m];
        A.xbf[m]  = (bf16*)(ws + take((size_t)BATCH * SEQ * XP[m] * sizeof(bf16)));
    }
    for (int m = 0; m < 3; ++m)
        for (int l = 0; l < 2; ++l)
            for (int d = 0; d < 2; ++d) {
                A.bih[m][l][d] = P[m][l][d][2];
                A.bhh[m][l][d] = P[m][l][d][3];
            }

    // 1) zero states (graph-capture safe)
    hipMemsetAsync(ws, 0, stateEnd, stream);

    // 2) pack weights -> bf16 Wcat (zero-padded x columns)
    for (int m = 0; m < 3; ++m)
        for (int l = 0; l < 2; ++l)
            for (int d = 0; d < 2; ++d) {
                const int kk  = (l == 0) ? K0[m] : 1536;
                const int Xl  = (l == 0) ? X[m] : 0;
                const int XPl = (l == 0) ? XP[m] : 0;
                const int inW = (l == 0) ? (X[m] + 1024) : 1024;
                const int total = 2048 * kk;
                pack_w_kernel<<<(total + 255) / 256, 256, 0, stream>>>(
                    A.wcat[m][l][d], P[m][l][d][0], P[m][l][d][1], Xl, XPl, inW, kk, total);
            }

    // 3) convert inputs -> padded bf16
    for (int m = 0; m < 3; ++m) {
        const int total = BATCH * SEQ * XP[m];
        cvt_x_kernel<<<(total + 255) / 256, 256, 0, stream>>>(
            A.xbf[m], (const float*)d_in[m], X[m], XP[m], total);
    }

    // 4) recurrent scan: 2 stages per step (stream ordering = grid sync),
    //    h ping-pong (read rp, write wp)
    for (int s = 0; s < SEQ; ++s) {
        const int rp = s & 1, wp = rp ^ 1;
        lstm_stage<<<dim3(32, 3, 2), 256, 0, stream>>>(A, s, 0, rp, wp);
        lstm_stage<<<dim3(32, 3, 2), 256, 0, stream>>>(A, s, 1, rp, wp);
    }

    // 5) decoder
    decode_kernel<<<1, 128, 0, stream>>>(A.st[0], A.st[1], A.st[2], decW, decb, (float*)d_out);
}
